// SelfAttention3D_13091060318672
// MI455X (gfx1250) — matrix operand order repaired and checked by reading, not compiled, not run on hardware
//
#include <hip/hip_runtime.h>

typedef _Float16 half_t;
typedef __attribute__((ext_vector_type(16))) _Float16 v16h;
typedef __attribute__((ext_vector_type(8)))  _Float16 v8h;
typedef __attribute__((ext_vector_type(8)))  float    v8f;
typedef __attribute__((ext_vector_type(4)))  unsigned int v4u;
typedef __attribute__((ext_vector_type(8)))  int v8i;
typedef __attribute__((ext_vector_type(4)))  int v4i;

#define B_    2
#define C_    256
#define C8_   32
#define N_    4096
#define NKEY  64   // keys staged in LDS per block iteration

#if __has_builtin(__builtin_amdgcn_tensor_load_to_lds) && \
    __has_builtin(__builtin_amdgcn_s_wait_tensorcnt)
#define USE_TDM 1
#endif

#if __has_builtin(__builtin_amdgcn_permlane16)
#define HAVE_PERMLANE 1
#define PERMLANE16(v, lo, hi) \
  __builtin_amdgcn_permlane16((v), (v), (lo), (hi), false, false)
#endif

static __device__ __forceinline__ v8f wmma_f16(v16h a, v16h b, v8f c) {
  // D(16x16,f32) = A(16x32,f16) x B(32x16,f16) + C
  return __builtin_amdgcn_wmma_f32_16x16x32_f16(
      /*neg_a=*/false, a, /*neg_b=*/false, b,
      /*c_mod=*/(short)0, c, /*reuse_a=*/false, /*reuse_b=*/false);
}

// Butterfly reductions across the 16-lane half (VALU v_permlane16_b32, no LDS).
static __device__ __forceinline__ float red16_max(float v) {
#ifdef HAVE_PERMLANE
  v = fmaxf(v, __uint_as_float(PERMLANE16(__float_as_uint(v), 0x67452301u, 0xEFCDAB89u))); // ^1
  v = fmaxf(v, __uint_as_float(PERMLANE16(__float_as_uint(v), 0x54761032u, 0xDCFE98BAu))); // ^2
  v = fmaxf(v, __uint_as_float(PERMLANE16(__float_as_uint(v), 0x32107654u, 0xBA98FEDCu))); // ^4
  v = fmaxf(v, __uint_as_float(PERMLANE16(__float_as_uint(v), 0xFEDCBA98u, 0x76543210u))); // ^8
#else
  v = fmaxf(v, __shfl_xor(v, 1, 32));
  v = fmaxf(v, __shfl_xor(v, 2, 32));
  v = fmaxf(v, __shfl_xor(v, 4, 32));
  v = fmaxf(v, __shfl_xor(v, 8, 32));
#endif
  return v;
}
static __device__ __forceinline__ float red16_sum(float v) {
#ifdef HAVE_PERMLANE
  v += __uint_as_float(PERMLANE16(__float_as_uint(v), 0x67452301u, 0xEFCDAB89u));
  v += __uint_as_float(PERMLANE16(__float_as_uint(v), 0x54761032u, 0xDCFE98BAu));
  v += __uint_as_float(PERMLANE16(__float_as_uint(v), 0x32107654u, 0xBA98FEDCu));
  v += __uint_as_float(PERMLANE16(__float_as_uint(v), 0xFEDCBA98u, 0x76543210u));
#else
  v += __shfl_xor(v, 1, 32);
  v += __shfl_xor(v, 2, 32);
  v += __shfl_xor(v, 4, 32);
  v += __shfl_xor(v, 8, 32);
#endif
  return v;
}

#ifdef USE_TDM
// Build a 2-D Tensor-DMA descriptor (ISA 8.3/8.4) and issue TENSOR_LOAD_TO_LDS.
// data_size = 2 bytes (f16). tile rows land contiguously in LDS.
// 6-arg builtin: (uint32x4 g0, int32x8 g1, int32x4 g2, int32x4 g3, int32x8 g4, i32 cpol)
static __device__ __forceinline__ void tdm_load_2d(
    const half_t* gaddr, half_t* laddr,
    unsigned tensor_d0, unsigned tensor_d1,
    unsigned tile_d0,   unsigned tile_d1, unsigned stride0)
{
  const unsigned long long ga = (unsigned long long)(uintptr_t)gaddr;
  const unsigned la = (unsigned)(uintptr_t)laddr;   // LDS byte offset
  v4u g0;
  g0[0] = 1u;                                        // count=1, user descriptor
  g0[1] = la;                                        // lds_addr
  g0[2] = (unsigned)(ga & 0xFFFFFFFFu);              // global_addr[31:0]
  g0[3] = (unsigned)((ga >> 32) & 0x01FFFFFFu)       // global_addr[56:32]
        | (2u << 30);                                // type = 2 ("image")
  v8i g1;
  g1[0] = (int)(1u << 16);                           // wg_mask=0, data_size=1 (2B)
  g1[1] = (int)((tensor_d0 & 0xFFFFu) << 16);        // tensor_dim0[15:0]
  g1[2] = (int)(((tensor_d0 >> 16) & 0xFFFFu)
        |  ((tensor_d1 & 0xFFFFu) << 16));           // dim0[31:16] | dim1[15:0]
  g1[3] = (int)(((tensor_d1 >> 16) & 0xFFFFu)
        |  ((tile_d0 & 0xFFFFu) << 16));             // dim1[31:16] | tile_dim0
  g1[4] = (int)(tile_d1 & 0xFFFFu);                  // tile_dim1 (tile_dim2=0)
  g1[5] = (int)stride0;                              // tensor_dim0_stride[31:0]
  g1[6] = 0;                                         // stride0[47:32] | stride1 lo
  g1[7] = 0;
  v4i g2 = {0,0,0,0};
  v4i g3 = {0,0,0,0};
  v8i g4 = {0,0,0,0,0,0,0,0};
  __builtin_amdgcn_tensor_load_to_lds(g0, g1, g2, g3, g4, 0);
}
#endif

// ---------------------------------------------------------------------------
// Projection: Y[o][n] = sum_c W[o][c] * X[c][n] + bias[o], emitted as f16.
// LAYOUT 0: out[b][n][COUT]  (Q/K — feeds WMMA A-matrix loads; packed b128 store)
// LAYOUT 1: out[b][COUT][N]  (V  — feeds WMMA B-matrix (V^T) loads)
// One wave per 16x16 output tile, fully unrolled 8 k-steps of K=32.
// ---------------------------------------------------------------------------
template<int LAYOUT, int COUT>
__global__ void __launch_bounds__(128) proj_kernel(
    const float* __restrict__ x, const float* __restrict__ W,
    const float* __restrict__ bias, half_t* __restrict__ out)
{
  const int lane = threadIdx.x & 31;
  const int wave = threadIdx.x >> 5;
  const int m = lane & 15;      // A row (o) / B col (n) within tile
  const int h = lane >> 4;      // lane half
  constexpr int rowTiles = COUT >> 4;
  constexpr int tilesPerBatch = rowTiles * (N_ >> 4);
  const int tile = blockIdx.x * 4 + wave;
  const int b  = tile / tilesPerBatch;
  const int tt = tile - b * tilesPerBatch;
  const int ot = tt / (N_ >> 4);
  const int nt = tt - ot * (N_ >> 4);
  const int obase = ot << 4, nbase = nt << 4;
  const float* xb = x + (size_t)b * C_ * N_;

  v8f acc = {0.f,0.f,0.f,0.f,0.f,0.f,0.f,0.f};
  #pragma unroll
  for (int kk = 0; kk < C_ / 32; ++kk) {
    // A-tile: W[obase+m][kk*32+...]; element e -> k=16h+e
    const float* wrow = W + (size_t)(obase + m) * C_ + kk * 32;
    v16h a;
    #pragma unroll
    for (int e = 0; e < 8; ++e) a[e]     = (half_t)wrow[16*h + e];
    #pragma unroll
    for (int e = 0; e < 8; ++e) a[8 + e] = (half_t)wrow[16*h + 8 + e];
    // B-tile: X[kk*32 + 16h + e][nbase+m]; element e -> k=16h+e, col=lane&15
    const float* xcol = xb + (size_t)(kk*32 + 16*h) * N_ + nbase + m;
    v16h bm;
    #pragma unroll
    for (int e = 0; e < 16; ++e) bm[e] = (half_t)xcol[(size_t)e * N_];
    acc = wmma_f16(a, bm, acc);
  }
  // C/D layout: vgpr r -> row M=r+8h, col N = lane&15
  const int n = nbase + m;
  if (LAYOUT == 0) {
    // o = obase + r + 8h is contiguous in r -> one 16B store per lane
    v8h pack;
    #pragma unroll
    for (int r = 0; r < 8; ++r)
      pack[r] = (half_t)(acc[r] + bias[obase + r + 8*h]);
    *(v8h*)(out + (size_t)b * N_ * COUT + (size_t)n * COUT + obase + 8*h) = pack;
  } else {
    #pragma unroll
    for (int r = 0; r < 8; ++r) {
      const int o = obase + r + 8*h;
      out[(size_t)b * COUT * N_ + (size_t)o * N_ + n] =
          (half_t)(acc[r] + bias[o]);
    }
  }
}

// ---------------------------------------------------------------------------
// Fused flash-attention + residual.
// Block = 256 threads = 8 waves. Waves 2w,2w+1 share query tile w (16 queries)
// and split the 256 output channels (8 c-tiles each -> 64 accumulator VGPRs).
// K/V staged in LDS 64 keys at a time (TDM); permlane16 softmax reductions;
// coalesced epilogue through LDS.
// ---------------------------------------------------------------------------
__global__ void __launch_bounds__(256) attn_kernel(
    const half_t* __restrict__ Qh, const half_t* __restrict__ Kh,
    const half_t* __restrict__ Vh, const float* __restrict__ x,
    const float* __restrict__ gamma, float* __restrict__ out)
{
  __shared__ __align__(16) half_t ldsK[NKEY * C8_];   // [key][ch]   4 KB
  __shared__ __align__(16) half_t ldsV[C_ * NKEY];    // [c][key]   32 KB (reused as f32 O stage)
  __shared__ __align__(16) half_t ldsP[8 * 16 * 32];  // per-wave P  8 KB

  const int lane = threadIdx.x & 31;
  const int wave = threadIdx.x >> 5;
  const int m = lane & 15;
  const int h = lane >> 4;
  const int b  = blockIdx.x >> 6;                       // 64 blocks per batch
  const int q0 = (blockIdx.x & 63) * 64;                // block's query base
  const int qb = q0 + (wave >> 1) * 16;                 // wave's query tile
  const int ct0 = (wave & 1) * 8;                       // this wave's c-tiles

  // Q A-tile: lane holds query row n=qb+m; element e -> ch 16h+e
  const half_t* qrow = Qh + ((size_t)b * N_ + qb + m) * C8_;
  const v8h qlo = *(const v8h*)(qrow + 16*h);
  const v8h qhi = *(const v8h*)(qrow + 16*h + 8);
  const v16h qa = __builtin_shufflevector(qlo, qhi,
      0,1,2,3,4,5,6,7,8,9,10,11,12,13,14,15);

  const v8f zf = {0.f,0.f,0.f,0.f,0.f,0.f,0.f,0.f};
  v8f oacc[8];
  #pragma unroll
  for (int t = 0; t < 8; ++t) oacc[t] = zf;
  float mstate[8], lstate[8];
  #pragma unroll
  for (int r = 0; r < 8; ++r) { mstate[r] = -1e30f; lstate[r] = 0.f; }

  const half_t* kb_g = Kh + (size_t)b * N_ * C8_;
  const half_t* vb_g = Vh + (size_t)b * C_ * N_;

  for (int mb = 0; mb < N_; mb += NKEY) {
    __syncthreads();   // previous stage fully consumed before overwrite
#ifdef USE_TDM
    if (wave == 0) {
      // K tile: 64 rows x 32 f16 out of the [4096][32] tensor (contiguous rows)
      tdm_load_2d(kb_g + (size_t)mb * C8_, ldsK,
                  /*tensor_d0=*/C8_, /*tensor_d1=*/N_,
                  /*tile_d0=*/C8_,   /*tile_d1=*/NKEY, /*stride0=*/C8_);
      // V tile: 256 rows x 64 f16 out of the [256][4096] tensor
      tdm_load_2d(vb_g + mb, ldsV,
                  /*tensor_d0=*/N_, /*tensor_d1=*/C_,
                  /*tile_d0=*/NKEY, /*tile_d1=*/C_, /*stride0=*/N_);
      __builtin_amdgcn_s_wait_tensorcnt(0);
    }
#else
    {
      const uint4* src = (const uint4*)(kb_g + (size_t)mb * C8_);
      uint4* dst = (uint4*)ldsK;
      for (int i = threadIdx.x; i < (NKEY * C8_ * 2) / 16; i += 256) dst[i] = src[i];
    }
    {
      uint4* dst = (uint4*)ldsV;
      for (int i = threadIdx.x; i < C_ * 8; i += 256) {
        const int c = i >> 3, j = i & 7;
        dst[i] = *(const uint4*)(vb_g + (size_t)c * N_ + mb + j * 8);
      }
    }
    if (mb + NKEY < N_) {
      __builtin_prefetch(kb_g + (size_t)(mb + NKEY) * C8_ + (size_t)threadIdx.x * 8, 0, 0);
      __builtin_prefetch(vb_g + (size_t)threadIdx.x * N_ + mb + NKEY, 0, 0);
    }
#endif
    __syncthreads();

    #pragma unroll
    for (int ms = 0; ms < NKEY; ms += 32) {
      // --- scores: two 16-key tiles, one WMMA each (K-dim = C8 = 32) ---
      const v16h kb0 = *(const v16h*)(ldsK + (ms + m) * C8_ + 16*h);
      const v16h kb1 = *(const v16h*)(ldsK + (ms + 16 + m) * C8_ + 16*h);
      v8f s0 = wmma_f16(qa, kb0, zf);
      v8f s1 = wmma_f16(qa, kb1, zf);

      // --- online softmax over the 32 new keys (permlane16 butterflies) ---
      float scale[8];
      #pragma unroll
      for (int r = 0; r < 8; ++r) {
        const float v = red16_max(fmaxf(s0[r], s1[r]));
        const float mnew = fmaxf(mstate[r], v);
        scale[r]  = __expf(mstate[r] - mnew);
        mstate[r] = mnew;
        s0[r] = __expf(s0[r] - mnew);
        s1[r] = __expf(s1[r] - mnew);
        lstate[r] = lstate[r] * scale[r] + red16_sum(s0[r] + s1[r]);
      }
      #pragma unroll
      for (int t = 0; t < 8; ++t)
        #pragma unroll
        for (int r = 0; r < 8; ++r) oacc[t][r] *= scale[r];

      // --- repack P (C/D layout) -> A layout via per-wave LDS scratch ---
      half_t* pw = ldsP + wave * (16 * 32);
      #pragma unroll
      for (int r = 0; r < 8; ++r) {
        const int n = r + 8*h;
        pw[n * 32 + m]      = (half_t)s0[r];
        pw[n * 32 + 16 + m] = (half_t)s1[r];
      }
      asm volatile("s_wait_dscnt 0" ::: "memory");  // cross-lane LDS RAW inside wave
      const v8h plo = *(const v8h*)(pw + m * 32 + 16*h);
      const v8h phi = *(const v8h*)(pw + m * 32 + 16*h + 8);
      const v16h pa = __builtin_shufflevector(plo, phi,
          0,1,2,3,4,5,6,7,8,9,10,11,12,13,14,15);

      // --- O += P x V^T : 8 c-tiles for this wave ---
      #pragma unroll
      for (int t = 0; t < 8; ++t) {
        const v16h vbm = *(const v16h*)(ldsV + ((ct0 + t) * 16 + m) * NKEY + ms + 16*h);
        oacc[t] = wmma_f16(pa, vbm, oacc[t]);
      }
    }
  }

  // --- epilogue: out = gamma * (O / l) + x, coalesced through LDS ---
  const float g = gamma[0];
  float inv[8];
  #pragma unroll
  for (int r = 0; r < 8; ++r) inv[r] = g / lstate[r];
  const float* xb = x + (size_t)b * C_ * N_;
  float* ob = out + (size_t)b * C_ * N_;
  float* ldsO = (float*)ldsV;   // reuse as f32 [128 channels][64 queries]

  #pragma unroll
  for (int p = 0; p < 2; ++p) {
    __syncthreads();
    if ((wave & 1) == p) {       // this wave's channels live in half p
      #pragma unroll
      for (int r = 0; r < 8; ++r) {
        const int n_local = (wave >> 1) * 16 + r + 8*h;   // 0..63
        #pragma unroll
        for (int t = 0; t < 8; ++t) {
          const int c_local = t * 16 + m;                 // 0..127 within half
          ldsO[c_local * 64 + n_local] = oacc[t][r] * inv[r];
        }
      }
    }
    __syncthreads();
    // 128 rows x 64 f32 streamed out coalesced; fuse residual add
    for (int i = threadIdx.x; i < 128 * 64; i += 256) {
      const int cl = i >> 6, nl = i & 63;
      const size_t idx = (size_t)(p * 128 + cl) * N_ + q0 + nl;
      ob[idx] = ldsO[i] + xb[idx];
    }
  }
}

// ---------------------------------------------------------------------------
extern "C" void kernel_launch(void* const* d_in, const int* in_sizes, int n_in,
                              void* d_out, int out_size, void* d_ws, size_t ws_size,
                              hipStream_t stream) {
  (void)in_sizes; (void)n_in; (void)out_size; (void)ws_size;
  const float* x     = (const float*)d_in[0];
  const float* wq    = (const float*)d_in[1];
  const float* bq    = (const float*)d_in[2];
  const float* wk    = (const float*)d_in[3];
  const float* bk    = (const float*)d_in[4];
  const float* wv    = (const float*)d_in[5];
  const float* bv    = (const float*)d_in[6];
  const float* gamma = (const float*)d_in[7];

  half_t* Qh = (half_t*)d_ws;                       // (B, N, 32)  512 KB
  half_t* Kh = Qh + (size_t)B_ * N_ * C8_;          // (B, N, 32)  512 KB
  half_t* Vh = Kh + (size_t)B_ * N_ * C8_;          // (B, 256, N) 4 MB

  // Q/K: 2 batches * 2 row-tiles * 256 col-tiles = 1024 tiles, 4 waves/block
  proj_kernel<0, C8_><<<256, 128, 0, stream>>>(x, wq, bq, Qh);
  proj_kernel<0, C8_><<<256, 128, 0, stream>>>(x, wk, bk, Kh);
  // V: 2 * 16 * 256 = 8192 tiles
  proj_kernel<1, C_><<<2048, 128, 0, stream>>>(x, wv, bv, Vh);
  // Attention: 2 batches * 64 blocks of 64 queries, 256 threads (8 waves)
  attn_kernel<<<128, 256, 0, stream>>>(Qh, Kh, Vh, x, gamma, (float*)d_out);
}
